// CustomLSTM_8899172237929
// MI455X (gfx1250) — matrix operand order repaired and checked by reading, not compiled, not run on hardware
//
#include <hip/hip_runtime.h>

// ---------------------------------------------------------------------------
// CustomLSTM for MI455X (gfx1250, wave32, WMMA bf16 16x16x32).
//
// Sizes: B=32, S=2048, I=256, H=256, W:(512,1024), b:(1024)
// out = (B,S,H) fp32 followed by two zero (1,B,H) tensors.
//
// Persistent 8-WG kernel; WG k owns hidden columns [32k,32k+32) of all four
// gates, with its 128KB bf16 weight slab pinned in LDS (WMMA-B fragment
// layout) for all 2048 steps.  The x-half of each step's GEMM is software-
// pipelined into the grid-barrier wait shadow (it has no dependence on h),
// so the serial recurrence path is only: h loads -> 8 chained wmmas ->
// activations -> cell update -> h store -> 8-WG atomic barrier.
// ---------------------------------------------------------------------------

typedef __attribute__((ext_vector_type(16))) __bf16 v16bf;
typedef __attribute__((ext_vector_type(8)))  float  v8f;

#define SEQ  2048
#define BAT  32
#define HID  256
#define N4H  1024

struct B32x8 { uint4 lo, hi; };

static __device__ __forceinline__ v16bf make_frag(uint4 lo, uint4 hi) {
  B32x8 t{lo, hi};
  return __builtin_bit_cast(v16bf, t);
}

static __device__ __forceinline__ unsigned short f2bf(float f) {
  unsigned u = __builtin_bit_cast(unsigned, f);
  u += 0x7FFFu + ((u >> 16) & 1u);           // round-to-nearest-even
  return (unsigned short)(u >> 16);
}

static __device__ __forceinline__ float sigmoidf(float x) {
  return 1.0f / (1.0f + __expf(-x));
}

// --------------------------- prep kernels ----------------------------------

__global__ void lstm_init_kernel(const float* __restrict__ h0,
                                 unsigned short* __restrict__ hbuf,
                                 unsigned* __restrict__ cnt,
                                 float* __restrict__ out) {
  int idx = blockIdx.x * 256 + threadIdx.x;
  if (idx < BAT * HID) hbuf[idx] = f2bf(h0[idx]);          // buffer 0
  if (idx < 2 * BAT * HID) out[(size_t)BAT * SEQ * HID + idx] = 0.0f;
  if (idx == 0) *cnt = 0u;
}

// x (B,S,I) fp32 -> xb [S][B][I] bf16 (A-fragment rows become contiguous)
__global__ void lstm_convert_x(const float* __restrict__ x,
                               unsigned short* __restrict__ xb) {
  int idx = blockIdx.x * 256 + threadIdx.x;       // output-linear
  int k = idx & 255;
  int b = (idx >> 8) & 31;
  int s = idx >> 13;
  xb[idx] = f2bf(x[((size_t)b * SEQ + s) * 256 + k]);
}

// W (512,1024) fp32 -> wswz bf16 in per-lane WMMA-B fragment layout, contiguous k order (same as A):
//   tile (kt, nt): 32 lanes x 8 dwords; lane L covers column nt*16+(L&15),
//   dword v holds K = kt*32 + (L>>4)*16 + 2v (lo) and +1 (hi).
__global__ void lstm_swizzle_w(const float* __restrict__ W,
                               unsigned* __restrict__ wswz) {
  int idx  = blockIdx.x * 256 + threadIdx.x;      // 32768 threads
  int lane = idx & 31;
  int nt   = (idx >> 5) & 63;
  int kt   = idx >> 11;
  int half = lane >> 4;
  int n    = nt * 16 + (lane & 15);
#pragma unroll
  for (int v = 0; v < 8; ++v) {
    int kk = kt * 32 + half * 16 + 2 * v;
    unsigned lo = f2bf(W[(size_t)kk * N4H + n]);
    unsigned hi = f2bf(W[(size_t)(kk + 1) * N4H + n]);
    wswz[(((size_t)kt * 64 + nt) * 32 + lane) * 8 + v] = lo | (hi << 16);
  }
}

// --------------------------- recurrent kernel ------------------------------

__global__ __launch_bounds__(512) void lstm_persistent(
    const unsigned short* __restrict__ xb,     // [S][B][256] bf16
    const uint4* __restrict__ wswz,            // swizzled W, 1MB
    unsigned short* __restrict__ hbuf,         // [2][B][256] bf16 double buf
    unsigned* __restrict__ cnt,                // grid barrier counter
    const float* __restrict__ c0,              // (B,H)
    const float* __restrict__ bias,            // (4H)
    float* __restrict__ out) {                 // (B,S,H)
  __shared__ uint4 w4[8192];                   // 128 KB: resident W slab
  __shared__ float gbuf[4][32][32];            // 16 KB: i/f/g/o activations
  __shared__ float cbuf[32][32];               // 4 KB: cell state slice

  const int tid   = threadIdx.x;
  const int wg    = blockIdx.x;                // 0..7
  const int lane  = tid & 31;
  const int wave  = tid >> 5;                  // 0..15
  const int mtile = wave & 1;                  // row-tile (b 0..15 / 16..31)
  const int ntl   = wave >> 1;                 // 0..7 local n-tile
  const int gate  = ntl >> 1;                  // 0..3
  const int half  = lane >> 4;

  // this wave's 16 output columns (fixed for the whole kernel)
  const int ncol  = gate * 256 + wg * 32 + (ntl & 1) * 16 + (lane & 15);
  const float bv  = bias[ncol];

  // stage W slab: 128 tiles x 64 uint4 from global swizzled W
#pragma unroll 4
  for (int i = tid; i < 8192; i += 512) {
    int t   = i >> 6;                          // local tile 0..127
    int off = i & 63;
    int kt  = t >> 3;
    int nl  = t & 7;
    int gnt = (nl >> 1) * 16 + wg * 2 + (nl & 1);   // global n-tile
    w4[i] = wswz[((size_t)kt * 64 + gnt) * 64 + off];
  }
  // init cell-state slice
  for (int i = tid; i < 1024; i += 512) {
    int b = i >> 5, j = i & 31;
    cbuf[b][j] = c0[b * HID + wg * 32 + j];
  }
  __syncthreads();

  const int bRow = mtile * 16 + (lane & 15);   // A-fragment row (batch idx)
  const int colL = (ntl & 1) * 16 + (lane & 15);

  uint4 xa[16];                                // x A-frags (next/current step)
  v8f accx;                                    // x-partial accumulator

  // ---- prologue: x-half of step 0 ----
  {
    const uint4* xr = (const uint4*)(xb + (size_t)bRow * 256);
#pragma unroll
    for (int kt = 0; kt < 8; ++kt) {
      xa[2 * kt]     = xr[kt * 4 + 2 * half];
      xa[2 * kt + 1] = xr[kt * 4 + 2 * half + 1];
    }
    accx = v8f{0.f, 0.f, 0.f, 0.f, 0.f, 0.f, 0.f, 0.f};
#pragma unroll
    for (int kt = 0; kt < 8; ++kt) {
      const uint4* bp = &w4[((kt * 8 + ntl) * 32 + lane) * 2];
      accx = __builtin_amdgcn_wmma_f32_16x16x32_bf16(
          false, make_frag(xa[2 * kt], xa[2 * kt + 1]),
          false, make_frag(bp[0], bp[1]), (short)0, accx, false, false);
    }
  }

  for (int s = 0; s < SEQ; ++s) {
    // ---- h-half: one load clause, then 8 chained wmmas onto accx ----
    uint4 ha[16];
    const uint4* hr =
        (const uint4*)(hbuf + (size_t)(s & 1) * BAT * HID + bRow * 256);
#pragma unroll
    for (int kt = 0; kt < 8; ++kt) {
      ha[2 * kt]     = hr[kt * 4 + 2 * half];
      ha[2 * kt + 1] = hr[kt * 4 + 2 * half + 1];
    }
    v8f acc = accx;
#pragma unroll
    for (int kt = 0; kt < 8; ++kt) {
      const uint4* bp = &w4[(((kt + 8) * 8 + ntl) * 32 + lane) * 2];
      acc = __builtin_amdgcn_wmma_f32_16x16x32_bf16(
          false, make_frag(ha[2 * kt], ha[2 * kt + 1]),
          false, make_frag(bp[0], bp[1]), (short)0, acc, false, false);
    }

    // ---- activation: z = tanh(acc + b); per-gate sigmoid/tanh ----
#pragma unroll
    for (int v = 0; v < 8; ++v) {
      int row = mtile * 16 + half * 8 + v;     // C-layout row for this VGPR
      float z = tanhf(acc[v] + bv);
      float a = (gate == 2) ? tanhf(z) : sigmoidf(z);
      gbuf[gate][row][colL] = a;
    }
    __syncthreads();

    // ---- issue next step's x A-frag loads (overlap elementwise VALU) ----
    if (s + 1 < SEQ) {
      const uint4* xr =
          (const uint4*)(xb + ((size_t)(s + 1) * BAT + bRow) * 256);
#pragma unroll
      for (int kt = 0; kt < 8; ++kt) {
        xa[2 * kt]     = xr[kt * 4 + 2 * half];
        xa[2 * kt + 1] = xr[kt * 4 + 2 * half + 1];
      }
    }

    // ---- cell update + h for this WG's 32 hidden cols (2 elems/thread) ----
    unsigned short* hnext = hbuf + (size_t)((s + 1) & 1) * BAT * HID;
#pragma unroll
    for (int e = 0; e < 2; ++e) {
      int idx = tid + e * 512;
      int b = idx >> 5, j = idx & 31;
      float iv = gbuf[0][b][j], fv = gbuf[1][b][j];
      float gv = gbuf[2][b][j], ov = gbuf[3][b][j];
      float c  = fv * cbuf[b][j] + iv * gv;
      cbuf[b][j] = c;
      float h  = tanhf(c) * ov;
      out[((size_t)b * SEQ + s) * HID + wg * 32 + j] = h;
      hnext[b * HID + wg * 32 + j] = f2bf(h);
    }

    // ---- x-half of step s+1: executes in the barrier-wait shadow ----
    if (s + 1 < SEQ) {
      accx = v8f{0.f, 0.f, 0.f, 0.f, 0.f, 0.f, 0.f, 0.f};
#pragma unroll
      for (int kt = 0; kt < 8; ++kt) {
        const uint4* bp = &w4[((kt * 8 + ntl) * 32 + lane) * 2];
        accx = __builtin_amdgcn_wmma_f32_16x16x32_bf16(
            false, make_frag(xa[2 * kt], xa[2 * kt + 1]),
            false, make_frag(bp[0], bp[1]), (short)0, accx, false, false);
      }
    }

    // ---- 8-WG grid barrier (all WGs co-resident: 8 WGs, 148KB LDS each) ----
    __threadfence();
    __syncthreads();
    if (tid == 0) {
      atomicAdd(cnt, 1u);
      unsigned target = 8u * (unsigned)(s + 1);
      while (__hip_atomic_load(cnt, __ATOMIC_RELAXED,
                               __HIP_MEMORY_SCOPE_AGENT) < target)
        __builtin_amdgcn_s_sleep(1);
    }
    __syncthreads();
    __threadfence();
  }
}

// --------------------------- launch ----------------------------------------

extern "C" void kernel_launch(void* const* d_in, const int* in_sizes, int n_in,
                              void* d_out, int out_size, void* d_ws, size_t ws_size,
                              hipStream_t stream) {
  const float* x    = (const float*)d_in[0];   // (B,S,I)
  const float* h0   = (const float*)d_in[1];   // (1,B,H)
  const float* c0   = (const float*)d_in[2];   // (1,B,H)
  const float* W    = (const float*)d_in[3];   // (512,1024)
  const float* bias = (const float*)d_in[4];   // (1024)
  float* out = (float*)d_out;

  char* ws = (char*)d_ws;
  unsigned short* xb   = (unsigned short*)(ws);               // 33,554,432 B
  unsigned*       wswz = (unsigned*)(ws + 33554432);          //  1,048,576 B
  unsigned short* hbuf = (unsigned short*)(ws + 34603008);    //     32,768 B
  unsigned*       cnt  = (unsigned*)(ws + 34635776);          //          4 B

  lstm_init_kernel<<<64, 256, 0, stream>>>(h0, hbuf, cnt, out);
  lstm_convert_x<<<65536, 256, 0, stream>>>(x, xb);
  lstm_swizzle_w<<<128, 256, 0, stream>>>(W, wswz);
  lstm_persistent<<<8, 512, 0, stream>>>(xb, (const uint4*)wswz, hbuf, cnt,
                                         c0, bias, out);
}
